// SplitNN_lca_71305047048727
// MI455X (gfx1250) — matrix operand order repaired and checked by reading, not compiled, not run on hardware
//
#include <hip/hip_runtime.h>
#include <hip/hip_bf16.h>

// ---------------------------------------------------------------------------
// CDNA5 (gfx1250) implementation of the LCA sparse-coding SplitNN pipeline.
// Heavy work (500-iteration LCA conv/convT loops, MLP GEMMs) runs on
// v_wmma_f32_16x16x32_bf16 with f32 accumulation; per-image LCA state lives
// entirely in LDS for the whole 500-iteration loop.
//
// Hot loops are chunk-outer / tile-inner with one accumulator per tile so
// consecutive WMMAs are independent (no WMMA->WMMA hazard NOPs) and LDS
// loads for tile j+1 overlap the WMMA of tile j.  Persistent state (u, hin)
// is pixel-major so epilogue access is contiguous b128 DS traffic.
// ---------------------------------------------------------------------------

typedef __attribute__((ext_vector_type(16))) __bf16 v16bf;
typedef __attribute__((ext_vector_type(8)))  __bf16 v8bf;
typedef __attribute__((ext_vector_type(8)))  float  v8f;
typedef __attribute__((ext_vector_type(4)))  float  v4f;

#define LCA_LAMBDA 0.4f
#define LCA_INVTAU 0.01f
#define LCA_ITERS  500

__device__ __forceinline__ v16bf zero_v16bf() {
    v16bf z;
#pragma unroll
    for (int j = 0; j < 16; ++j) z[j] = (__bf16)0.0f;
    return z;
}

__device__ __forceinline__ v8f zero_v8f() {
    v8f z = {0.f, 0.f, 0.f, 0.f, 0.f, 0.f, 0.f, 0.f};
    return z;
}

__device__ __forceinline__ v8f wmma_bf16(v16bf a, v16bf b, v8f c) {
    return __builtin_amdgcn_wmma_f32_16x16x32_bf16(false, a, false, b,
                                                   (short)0, c, false, false);
}

// A-fragment per-lane K ordering for 16-bit 16x32 A tiles:
// lane half h: element e -> k = 16h+e
__device__ __forceinline__ int afrag_k(int h, int e) {
    return 16 * h + e;
}

__device__ __forceinline__ float relu_l(float v) {
    return v > LCA_LAMBDA ? v - LCA_LAMBDA : 0.0f;
}

// ===========================================================================
// LCA layer 1:  x[1ch] -> u[16ch], 28x28, 5x5 dict, 500 iterations.
// One block per image. conv (1->16) = single 32-K-chunk WMMA (k = tap).
// convT (16->1) = 13 chunks (k = tap*16 + ci), only M row 0 meaningful.
// u is pixel-major [784][16] for contiguous epilogue access.
// ===========================================================================
__global__ __launch_bounds__(256)
void lca1_kernel(const float* __restrict__ x_g,
                 const float* __restrict__ Wd1,      // [16,1,5,5]
                 float* __restrict__ a1_out)         // [16,16,28,28]
{
    extern __shared__ char sm1[];
    float*  u   = (float*)sm1;                        // [784][16] f32 (50176 B)
    float*  xin = (float*)(sm1 + 50176);              // 784 f32      (3136 B)
    __bf16* ah  = (__bf16*)(sm1 + 53312);             // [32][32][16] (32768 B)
    __bf16* eh  = (__bf16*)(sm1 + 86080);             // [32][32]     (2048 B)
    __bf16* WcA = (__bf16*)(sm1 + 88128);             // [32][16]     (1024 B)
    __bf16* WtA = (__bf16*)(sm1 + 89152);             // [13][32][16] (13312 B)

    const int b    = blockIdx.x;
    const int tid  = threadIdx.x;
    const int lane = tid & 31;
    const int wv   = tid >> 5;
    const int kh   = lane >> 4;   // k-half within wave
    const int l16  = lane & 15;

    // ---- init ----
    for (int i = tid; i < 784 * 16; i += 256) u[i] = 0.0f;
    for (int i = tid; i < 784; i += 256) xin[i] = x_g[b * 784 + i];
    for (int i = tid; i < 32 * 32 * 16; i += 256) ah[i] = (__bf16)0.0f;
    for (int i = tid; i < 32 * 32; i += 256) eh[i] = (__bf16)0.0f;
    // conv weights, A layout: A[m][k=tap] = Wd1[m*25+tap]
    for (int i = tid; i < 32 * 16; i += 256) {
        int e = i & 15, ln = i >> 4;
        int m = ln & 15, h = ln >> 4;
        int kl = afrag_k(h, e);
        WcA[i] = (__bf16)((kl < 25) ? Wd1[m * 25 + kl] : 0.0f);
    }
    // convT weights: A[m][k=tap*16+ci] = (m==0) ? Wd1[ci*25+tap] : 0
    for (int i = tid; i < 13 * 32 * 16; i += 256) {
        int e = i & 15, ln = (i >> 4) & 31, kc = i >> 9;
        int m = ln & 15, h = ln >> 4;
        int k = 32 * kc + afrag_k(h, e);
        int tap = k >> 4, ci = k & 15;
        WtA[i] = (__bf16)((m == 0 && tap < 25) ? Wd1[ci * 25 + tap] : 0.0f);
    }
    __syncthreads();

    // per-wave tile slots: nt = wv + 8*j (j < 7), 49 tiles total
    int sp[7], sy[7], sx[7];
#pragma unroll
    for (int j = 0; j < 7; ++j) {
        int p = (wv + 8 * j) * 16 + l16;
        sp[j] = p; sy[j] = p / 28; sx[j] = p % 28;
    }

    // conv A fragment is iteration-invariant -> keep in registers forever
    const v16bf afc = *(const v16bf*)(WcA + lane * 16);

    for (int it = 0; it < LCA_ITERS; ++it) {
        // phase 1: recon = convT(a)  (chunk-outer, one acc per tile)
        v8f acc[7];
#pragma unroll
        for (int j = 0; j < 7; ++j) acc[j] = zero_v8f();

        for (int kc = 0; kc < 13; ++kc) {
            v16bf af = *(const v16bf*)(WtA + (kc * 32 + lane) * 16);
            int t = 2 * kc + kh;
            bool tv = t < 25;
            int koff = ((t / 5) * 32 + (t % 5)) * 16;    // halo offset (elems)
#pragma unroll
            for (int j = 0; j < 7; ++j) {
                if (wv + 8 * j < 49) {
                    v16bf bf = tv
                        ? *(const v16bf*)(ah + ((sy[j] + 4) * 32 + (sx[j] + 4)) * 16 - koff)
                        : zero_v16bf();
                    acc[j] = wmma_bf16(af, bf, acc[j]);
                }
            }
        }
#pragma unroll
        for (int j = 0; j < 7; ++j) {
            if (wv + 8 * j < 49 && kh == 0) {   // C row M=0: VGPR0, lanes 0-15
                eh[(sy[j] + 2) * 32 + (sx[j] + 2)] = (__bf16)(xin[sp[j]] - acc[j][0]);
            }
        }
        __syncthreads();

        // phase 2: drive = conv(e); u += (drive + a - u)/tau; write a-halo
#pragma unroll
        for (int j = 0; j < 7; ++j) {
            if (wv + 8 * j < 49) {
                v16bf bf;
#pragma unroll
                for (int jj = 0; jj < 16; ++jj) {
                    int t = 16 * kh + jj;
                    bf[jj] = (t < 25) ? eh[(sy[j] + t / 5) * 32 + (sx[j] + t % 5)]
                                      : (__bf16)0.0f;
                }
                v8f dr = wmma_bf16(afc, bf, zero_v8f());
                // contiguous pixel-major update: ch m = 8*kh + r
                float* up = u + sp[j] * 16 + 8 * kh;
                v4f u0 = *(const v4f*)up;
                v4f u1 = *(const v4f*)(up + 4);
                v8bf av;
#pragma unroll
                for (int r = 0; r < 4; ++r) {
                    float a = relu_l(u0[r]);
                    u0[r] = u0[r] + (dr[r] + a - u0[r]) * LCA_INVTAU;
                    av[r] = (__bf16)relu_l(u0[r]);
                }
#pragma unroll
                for (int r = 0; r < 4; ++r) {
                    float a = relu_l(u1[r]);
                    u1[r] = u1[r] + (dr[4 + r] + a - u1[r]) * LCA_INVTAU;
                    av[4 + r] = (__bf16)relu_l(u1[r]);
                }
                *(v4f*)up = u0;
                *(v4f*)(up + 4) = u1;
                *(v8bf*)(ah + ((sy[j] + 2) * 32 + (sx[j] + 2)) * 16 + 8 * kh) = av;
            }
        }
        __syncthreads();
    }

    for (int i = tid; i < 16 * 784; i += 256) {
        int m = i / 784, p = i % 784;
        float uu = u[p * 16 + m];
        a1_out[b * 16 * 784 + i] = uu > LCA_LAMBDA ? uu - LCA_LAMBDA : 0.0f;
    }
}

// ===========================================================================
// LCA layer 2: h[16ch] -> u[28ch], the dominant 281 GFLOP loop.
// One block per image; all state resident in LDS for 500 iterations.
// convT: K = tap*32+ci (25 chunks);  conv: K = tap*16+ci (13 chunks, 2 taps)
// conv shares each B fragment between both M-tiles (2 WMMAs per B load).
// u is pixel-major [784][32] (channels padded to 32 => no edge guards:
// A rows >=28 are zero-packed so padded u/a-halo channels stay zero).
// ===========================================================================
__global__ __launch_bounds__(256)
void lca2_kernel(const float* __restrict__ hin_g,    // bn1 out [16,16,28,28]
                 const float* __restrict__ Wd2,      // [28,16,5,5]
                 float* __restrict__ a2_out)         // [16,28,28,28]
{
    extern __shared__ char sm2[];
    float*  u     = (float*)sm2;                      // [784][32] f32 (100352 B)
    float*  hin   = (float*)(sm2 + 100352);           // [784][16] f32 (50176 B)
    __bf16* ahalo = (__bf16*)(sm2 + 150528);          // [32][32][32] (65536 B)
    __bf16* ehalo = (__bf16*)(sm2 + 216064);          // [32][32][16] (32768 B)
    __bf16* WcA   = (__bf16*)(sm2 + 248832);          // [2][13][32][16] 26624 B
    __bf16* WtA   = (__bf16*)(sm2 + 275456);          // [25][32][16]    25600 B

    const int b    = blockIdx.x;
    const int tid  = threadIdx.x;
    const int lane = tid & 31;
    const int wv   = tid >> 5;
    const int kh   = lane >> 4;
    const int l16  = lane & 15;

    __builtin_prefetch(Wd2, 0, 3);

    // ---- init ----
    for (int i = tid; i < 784 * 32; i += 256) u[i] = 0.0f;
    for (int i = tid; i < 16 * 784; i += 256) {     // transpose to pixel-major
        int c = i / 784, p = i % 784;
        hin[p * 16 + c] = hin_g[b * 16 * 784 + i];
    }
    for (int i = tid; i < 32 * 32 * 32; i += 256) ahalo[i] = (__bf16)0.0f;
    for (int i = tid; i < 32 * 32 * 16; i += 256) ehalo[i] = (__bf16)0.0f;
    // conv weights: A[o][k=tap*16+ci] = Wd2[o][ci][ky][kx]
    for (int i = tid; i < 2 * 13 * 32 * 16; i += 256) {
        int e = i & 15, ln = (i >> 4) & 31;
        int kc = (i >> 9) % 13, mt = (i >> 9) / 13;
        int m = ln & 15, h = ln >> 4;
        int k = 32 * kc + afrag_k(h, e);
        int t = k >> 4, ci = k & 15;
        int o = mt * 16 + m;
        float w = 0.0f;
        if (o < 28 && t < 25)
            w = Wd2[((o * 16 + ci) * 5 + (t / 5)) * 5 + (t % 5)];
        WcA[i] = (__bf16)w;
    }
    // convT weights: A[i][k=tap*32+o] = Wd2[o][i][ky][kx]  (o<28, pad to 32)
    for (int i = tid; i < 25 * 32 * 16; i += 256) {
        int e = i & 15, ln = (i >> 4) & 31, tap = i >> 9;
        int m = ln & 15, h = ln >> 4;
        int o = afrag_k(h, e);  // 0..31
        float w = 0.0f;
        if (o < 28)
            w = Wd2[((o * 16 + m) * 5 + (tap / 5)) * 5 + (tap % 5)];
        WtA[i] = (__bf16)w;
    }
    __syncthreads();

    // per-wave tile slots: nt = wv + 8*j (j < 7), 49 tiles total
    int sp[7], sy[7], sx[7];
#pragma unroll
    for (int j = 0; j < 7; ++j) {
        int p = (wv + 8 * j) * 16 + l16;
        sp[j] = p; sy[j] = p / 28; sx[j] = p % 28;
    }

    for (int it = 0; it < LCA_ITERS; ++it) {
        // ---- phase 1: recon = convT(a) [16ch]; e = hin - recon -> ehalo ----
        {
            v8f acc[7];
#pragma unroll
            for (int j = 0; j < 7; ++j) acc[j] = zero_v8f();

            for (int tap = 0; tap < 25; ++tap) {
                v16bf af = *(const v16bf*)(WtA + (tap * 32 + lane) * 16);
                int koff = ((tap / 5) * 32 + (tap % 5)) * 32;   // halo offset
#pragma unroll
                for (int j = 0; j < 7; ++j) {
                    if (wv + 8 * j < 49) {
                        v16bf bf = *(const v16bf*)(ahalo +
                            ((sy[j] + 4) * 32 + (sx[j] + 4)) * 32 + 16 * kh - koff);
                        acc[j] = wmma_bf16(af, bf, acc[j]);
                    }
                }
            }
#pragma unroll
            for (int j = 0; j < 7; ++j) {
                if (wv + 8 * j < 49) {
                    // ch ich = 8*kh + r : contiguous in pixel-major hin/ehalo
                    const float* hp = hin + sp[j] * 16 + 8 * kh;
                    v4f h0 = *(const v4f*)hp;
                    v4f h1 = *(const v4f*)(hp + 4);
                    v8bf ev;
#pragma unroll
                    for (int r = 0; r < 4; ++r) ev[r] = (__bf16)(h0[r] - acc[j][r]);
#pragma unroll
                    for (int r = 0; r < 4; ++r) ev[4 + r] = (__bf16)(h1[r] - acc[j][4 + r]);
                    *(v8bf*)(ehalo + ((sy[j] + 2) * 32 + (sx[j] + 2)) * 16 + 8 * kh) = ev;
                }
            }
        }
        __syncthreads();

        // ---- phase 2: drive = conv(e) [28ch pad 32]; u update; a-halo ----
        {
            v8f acc0[7], acc1[7];
#pragma unroll
            for (int j = 0; j < 7; ++j) { acc0[j] = zero_v8f(); acc1[j] = zero_v8f(); }

            for (int kc = 0; kc < 13; ++kc) {
                v16bf a0 = *(const v16bf*)(WcA + (kc * 32 + lane) * 16);
                v16bf a1 = *(const v16bf*)(WcA + ((13 + kc) * 32 + lane) * 16);
                int t = 2 * kc + kh;
                bool tv = t < 25;
                int toff = ((t / 5) * 32 + (t % 5)) * 16;
#pragma unroll
                for (int j = 0; j < 7; ++j) {
                    if (wv + 8 * j < 49) {
                        v16bf bf = tv
                            ? *(const v16bf*)(ehalo + (sy[j] * 32 + sx[j]) * 16 + toff)
                            : zero_v16bf();
                        acc0[j] = wmma_bf16(a0, bf, acc0[j]);
                        acc1[j] = wmma_bf16(a1, bf, acc1[j]);
                    }
                }
            }
#pragma unroll
            for (int j = 0; j < 7; ++j) {
                if (wv + 8 * j < 49) {
                    int hb = ((sy[j] + 2) * 32 + (sx[j] + 2)) * 32;
#pragma unroll
                    for (int tmt = 0; tmt < 2; ++tmt) {  // M-tiles: ch 0..15, 16..31
                        v8f dr = (tmt == 0) ? acc0[j] : acc1[j];
                        float* up = u + sp[j] * 32 + tmt * 16 + 8 * kh;
                        v4f u0 = *(const v4f*)up;
                        v4f u1 = *(const v4f*)(up + 4);
                        v8bf av;
#pragma unroll
                        for (int r = 0; r < 4; ++r) {
                            float a = relu_l(u0[r]);
                            u0[r] = u0[r] + (dr[r] + a - u0[r]) * LCA_INVTAU;
                            av[r] = (__bf16)relu_l(u0[r]);
                        }
#pragma unroll
                        for (int r = 0; r < 4; ++r) {
                            float a = relu_l(u1[r]);
                            u1[r] = u1[r] + (dr[4 + r] + a - u1[r]) * LCA_INVTAU;
                            av[4 + r] = (__bf16)relu_l(u1[r]);
                        }
                        *(v4f*)up = u0;
                        *(v4f*)(up + 4) = u1;
                        *(v8bf*)(ahalo + hb + tmt * 16 + 8 * kh) = av;
                    }
                }
            }
        }
        __syncthreads();
    }

    for (int i = tid; i < 28 * 784; i += 256) {
        int m = i / 784, p = i % 784;
        float uu = u[p * 32 + m];
        a2_out[b * 28 * 784 + i] = uu > LCA_LAMBDA ? uu - LCA_LAMBDA : 0.0f;
    }
}

// ===========================================================================
// BatchNorm (train-mode batch stats over N,H,W) — one block per channel.
// ===========================================================================
__global__ __launch_bounds__(256)
void bn_kernel(const float* __restrict__ in, float* __restrict__ out,
               const float* __restrict__ g, const float* __restrict__ bt, int C)
{
    __shared__ float r1[256];
    __shared__ float r2[256];
    const int c = blockIdx.x;
    const int tid = threadIdx.x;
    float s = 0.f, s2 = 0.f;
    for (int i = tid; i < 16 * 784; i += 256) {
        int bb = i / 784, p = i % 784;
        float v = in[(bb * C + c) * 784 + p];
        s += v; s2 += v * v;
    }
    r1[tid] = s; r2[tid] = s2;
    __syncthreads();
    for (int o = 128; o > 0; o >>= 1) {
        if (tid < o) { r1[tid] += r1[tid + o]; r2[tid] += r2[tid + o]; }
        __syncthreads();
    }
    float mean = r1[0] * (1.0f / 12544.0f);
    float var  = r2[0] * (1.0f / 12544.0f) - mean * mean;
    float scale = g[c] * rsqrtf(var + 1e-5f);
    float shift = bt[c] - mean * scale;
    for (int i = tid; i < 16 * 784; i += 256) {
        int bb = i / 784, p = i % 784;
        int idx = (bb * C + c) * 784 + p;
        out[idx] = in[idx] * scale + shift;
    }
}

// ===========================================================================
// prep: convert W2 [500,14000] f32 -> bf16 with row stride 14016 (zero pad),
// and zero the pad columns of the stage-2 A matrix (out1).
// ===========================================================================
__global__ void prep_kernel(const float* __restrict__ W2,
                            __bf16* __restrict__ W2b,
                            __bf16* __restrict__ out1)
{
    const int stride = gridDim.x * blockDim.x;
    for (int i = blockIdx.x * blockDim.x + threadIdx.x; i < 500 * 14016; i += stride) {
        int n = i / 14016, k = i % 14016;
        W2b[i] = (__bf16)((k < 14000) ? W2[n * 14000 + k] : 0.0f);
    }
    for (int j = blockIdx.x * blockDim.x + threadIdx.x; j < 448 * 16; j += stride) {
        int r = j / 16, k = j % 16;
        out1[r * 14016 + 14000 + k] = (__bf16)0.0f;
    }
}

// ===========================================================================
// MLP stage 1: out[bch, o] = relu(h2[bch,:28] . W1[o,:28] + b1[o]).
// Emits bf16 directly into the [448 x 14016] stage-2 A layout.
// One block per 16-row M tile (784 tiles); 8 waves x 4 N tiles (N=512 pad).
// ===========================================================================
__global__ __launch_bounds__(256)
void mlp1_kernel(const float* __restrict__ h2, const float* __restrict__ W1,
                 const float* __restrict__ b1, __bf16* __restrict__ out1)
{
    __shared__ __bf16 Ar[16 * 32];
    __shared__ __bf16 Bw[512 * 32];
    const int mt = blockIdx.x;
    const int tid = threadIdx.x;
    const int lane = tid & 31, wv = tid >> 5, kh = lane >> 4, l16 = lane & 15;

    for (int i = tid; i < 16 * 32; i += 256) {
        int m = i >> 5, k = i & 31;
        Ar[i] = (__bf16)((k < 28) ? h2[(mt * 16 + m) * 28 + k] : 0.0f);
    }
    for (int i = tid; i < 512 * 32; i += 256) {
        int n = i >> 5, k = i & 31;
        Bw[i] = (__bf16)((n < 500 && k < 28) ? W1[n * 28 + k] : 0.0f);
    }
    __syncthreads();

    v16bf af;
#pragma unroll
    for (int e = 0; e < 16; ++e) af[e] = Ar[l16 * 32 + afrag_k(kh, e)];

    for (int q = 0; q < 4; ++q) {
        int nt = wv * 4 + q;
        v16bf bf = *(const v16bf*)(Bw + (nt * 16 + l16) * 32 + 16 * kh);
        v8f acc = wmma_bf16(af, bf, zero_v8f());
        int o = nt * 16 + l16;
        if (o < 500) {
            float bias = b1[o];
#pragma unroll
            for (int r = 0; r < 8; ++r) {
                int rowg = mt * 16 + r + 8 * kh;     // bch index < 12544
                float v = acc[r] + bias;
                v = v > 0.f ? v : 0.f;
                int rr = rowg / 28;                  // stage-2 row (b*28+c)
                int cc = (rowg % 28) * 500 + o;      // stage-2 col (h*500+o)
                out1[rr * 14016 + cc] = (__bf16)v;
            }
        }
    }
}

// ===========================================================================
// MLP stage 2: [448 x 14016(bf16)] x W2b^T -> relu -> [448 x 500] f32.
// One block per M tile (28); 8 waves x 4 N tiles; 438 K chunks.
// ===========================================================================
__global__ __launch_bounds__(256)
void mlp2_kernel(const __bf16* __restrict__ A, const __bf16* __restrict__ Bm,
                 const float* __restrict__ b2, float* __restrict__ out2)
{
    const int mt = blockIdx.x;
    const int tid = threadIdx.x;
    const int lane = tid & 31, wv = tid >> 5, kh = lane >> 4, l16 = lane & 15;
    const int rowg = mt * 16 + l16;                  // < 448
    const __bf16* arow = A + rowg * 14016;

    v8f acc[4];
#pragma unroll
    for (int q = 0; q < 4; ++q) acc[q] = zero_v8f();

    for (int kc = 0; kc < 438; ++kc) {
        v16bf af;
        const __bf16* p0 = arow + kc * 32 + 16 * kh;
#pragma unroll
        for (int e = 0; e < 8; ++e) af[e] = p0[e];
#pragma unroll
        for (int e = 0; e < 8; ++e) af[8 + e] = p0[8 + e];
#pragma unroll
        for (int q = 0; q < 4; ++q) {
            int n = (wv * 4 + q) * 16 + l16;
            int nc = n < 500 ? n : 499;              // clamped; junk cols dropped
            v16bf bf = *(const v16bf*)(Bm + nc * 14016 + kc * 32 + 16 * kh);
            acc[q] = wmma_bf16(af, bf, acc[q]);
        }
    }
#pragma unroll
    for (int q = 0; q < 4; ++q) {
        int n = (wv * 4 + q) * 16 + l16;
        if (n < 500) {
            float bias = b2[n];
#pragma unroll
            for (int r = 0; r < 8; ++r) {
                int m = mt * 16 + r + 8 * kh;
                float v = acc[q][r] + bias;
                out2[m * 500 + n] = v > 0.f ? v : 0.f;
            }
        }
    }
}

// ===========================================================================
// Head: logits = out2 @ W3^T + b3, softmax.  One wave per row.
// ===========================================================================
__global__ void head_kernel(const float* __restrict__ out2,
                            const float* __restrict__ W3,
                            const float* __restrict__ b3,
                            float* __restrict__ out)
{
    const int r = blockIdx.x;       // 0..447
    const int lane = threadIdx.x;   // 32 lanes
    float acc[10];
#pragma unroll
    for (int j = 0; j < 10; ++j) acc[j] = 0.f;
    for (int k = lane; k < 500; k += 32) {
        float h = out2[r * 500 + k];
#pragma unroll
        for (int j = 0; j < 10; ++j) acc[j] += h * W3[j * 500 + k];
    }
#pragma unroll
    for (int j = 0; j < 10; ++j)
        for (int off = 16; off > 0; off >>= 1)
            acc[j] += __shfl_down(acc[j], off, 32);
    if (lane == 0) {
        float mx = -1e30f;
#pragma unroll
        for (int j = 0; j < 10; ++j) { acc[j] += b3[j]; mx = acc[j] > mx ? acc[j] : mx; }
        float s = 0.f;
#pragma unroll
        for (int j = 0; j < 10; ++j) { acc[j] = __expf(acc[j] - mx); s += acc[j]; }
        float inv = 1.0f / s;
#pragma unroll
        for (int j = 0; j < 10; ++j) out[r * 10 + j] = acc[j] * inv;
    }
}

// ===========================================================================
// launch
// ===========================================================================
extern "C" void kernel_launch(void* const* d_in, const int* in_sizes, int n_in,
                              void* d_out, int out_size, void* d_ws, size_t ws_size,
                              hipStream_t stream) {
    const float* x     = (const float*)d_in[0];
    const float* Wd1   = (const float*)d_in[1];
    const float* Wd2   = (const float*)d_in[2];
    const float* bn1_g = (const float*)d_in[3];
    const float* bn1_b = (const float*)d_in[4];
    const float* bn2_g = (const float*)d_in[5];
    const float* bn2_b = (const float*)d_in[6];
    const float* W1    = (const float*)d_in[7];
    const float* b1    = (const float*)d_in[8];
    const float* W2    = (const float*)d_in[9];
    const float* b2    = (const float*)d_in[10];
    const float* W3    = (const float*)d_in[11];
    const float* b3    = (const float*)d_in[12];

    char* ws = (char*)d_ws;
    float*  a1   = (float*)(ws + 0);          // 16*16*784 f32  (802816 B)
    float*  h1   = (float*)(ws + 802816);     // 16*16*784 f32
    float*  a2   = (float*)(ws + 1605632);    // 16*28*784 f32  (1404928 B)
    float*  h2   = (float*)(ws + 3010560);    // 16*28*784 f32
    __bf16* out1 = (__bf16*)(ws + 4415488);   // 448*14016 bf16 (12558336 B)
    __bf16* W2b  = (__bf16*)(ws + 16973824);  // 500*14016 bf16 (14016000 B)
    float*  out2 = (float*)(ws + 30989824);   // 448*500 f32    (896000 B)

    // prep (independent of LCA chain)
    prep_kernel<<<256, 256, 0, stream>>>(W2, W2b, out1);

    // LCA layer 1 + BN1
    lca1_kernel<<<16, 256, 102464, stream>>>(x, Wd1, a1);
    bn_kernel<<<16, 256, 0, stream>>>(a1, h1, bn1_g, bn1_b, 16);

    // LCA layer 2 + BN2  (dominant: 500 iters, all state in LDS, WMMA convs)
    lca2_kernel<<<16, 256, 301056, stream>>>(h1, Wd2, a2);
    bn_kernel<<<28, 256, 0, stream>>>(a2, h2, bn2_g, bn2_b, 28);

    // MLP
    mlp1_kernel<<<784, 256, 0, stream>>>(h2, W1, b1, out1);
    mlp2_kernel<<<28, 256, 0, stream>>>(out1, W2b, b2, out2);
    head_kernel<<<448, 32, 0, stream>>>(out2, W3, b3, (float*)d_out);
}
